// MultiHeadAttention_31035433681576
// MI455X (gfx1250) — matrix operand order repaired and checked by reading, not compiled, not run on hardware
//
#include <hip/hip_runtime.h>
#include <hip/hip_bf16.h>
#include <stdint.h>

#define BB 4
#define LLQ 1024
#define DD 512
#define HH 8
#define NDH 4096  // H*D

typedef __bf16 bf16_t;
typedef __attribute__((ext_vector_type(16))) __bf16 v16bf;
typedef __attribute__((ext_vector_type(8)))  float  v8f;

// ---------- bf16 helpers (RNE) ----------
__device__ __forceinline__ bf16_t f2bf(float f) {
  union { float f; uint32_t u; } a; a.f = f;
  uint32_t r = a.u + 0x7fffu + ((a.u >> 16) & 1u);
  union { uint16_t s; bf16_t b; } o; o.s = (uint16_t)(r >> 16);
  return o.b;
}
__device__ __forceinline__ float bf2f(bf16_t b) {
  union { uint16_t s; bf16_t b; } i; i.b = b;
  union { uint32_t u; float f; } o; o.u = ((uint32_t)i.s) << 16;
  return o.f;
}

// ---------- WMMA fragment loads (ISA 7.12.2 layouts, wave32) ----------
// A (16xK slab, row-major, K contiguous): lane<16 holds row=lane K[0..7],K[16..23];
// lane>=16 holds row=lane-16 ... K[8..15],K[24..31].
__device__ __forceinline__ v16bf load_frag_a(const bf16_t* p16, int ld, int lane) {
  const bf16_t* p = p16 + (size_t)(lane & 15) * ld + ((lane >> 4) << 3);
  union { uint4 u[2]; v16bf v; } x;
  x.u[0] = *(const uint4*)p;
  x.u[1] = *(const uint4*)(p + 16);
  return x.v;
}
// B (N rows, K contiguous == K x 16 fragment): lane<16 col=lane K[0..7],K[16..23];
// lane>=16 col=lane-16 K[8..15],K[24..31].
__device__ __forceinline__ v16bf load_frag_b(const bf16_t* p16, int ld, int lane) {
  const bf16_t* p = p16 + (size_t)(lane & 15) * ld + ((lane >> 4) << 3);
  union { uint4 u[2]; v16bf v; } x;
  x.u[0] = *(const uint4*)p;
  x.u[1] = *(const uint4*)(p + 16);
  return x.v;
}

// ---------- wave-level 64x64 NT GEMM, bf16 in / f32 accumulate ----------
__device__ __forceinline__ void gemm64x64(const bf16_t* __restrict__ A, int lda,
                                          const bf16_t* __restrict__ Bm, int ldb,
                                          int K, int lane, v8f acc[4][4]) {
#pragma unroll
  for (int i = 0; i < 4; i++)
#pragma unroll
    for (int j = 0; j < 4; j++)
#pragma unroll
      for (int r = 0; r < 8; r++) acc[i][j][r] = 0.f;

  for (int kk = 0; kk < K; kk += 32) {
    v16bf af[4], bfr[4];
#pragma unroll
    for (int i = 0; i < 4; i++)
      af[i] = load_frag_a(A + (size_t)(i * 16) * lda + kk, lda, lane);
#pragma unroll
    for (int j = 0; j < 4; j++)
      bfr[j] = load_frag_b(Bm + (size_t)(j * 16) * ldb + kk, ldb, lane);
    if (kk + 32 < K) {
      __builtin_prefetch(A + (size_t)(lane & 15) * lda + kk + 32);
      __builtin_prefetch(Bm + (size_t)(lane & 15) * ldb + kk + 32);
    }
#pragma unroll
    for (int i = 0; i < 4; i++)
#pragma unroll
      for (int j = 0; j < 4; j++)
        acc[i][j] = __builtin_amdgcn_wmma_f32_16x16x32_bf16(
            false, af[i], false, bfr[j], (short)0, acc[i][j], false, false);
  }
}

// ---------- fp32 -> bf16 convert ----------
__global__ void cvt_kernel(const float* __restrict__ in, bf16_t* __restrict__ out, int n) {
  int i = blockIdx.x * 256 + threadIdx.x;
  if (i < n) out[i] = f2bf(in[i]);
}

// ---------- QKV projection: Y[b,h,l,e] (or transposed [b,h,e,l]) = X @ W^T ----------
__global__ __launch_bounds__(256) void proj_kernel(const bf16_t* __restrict__ X,
                                                   const bf16_t* __restrict__ W,
                                                   bf16_t* __restrict__ Y, int vtrans) {
  int lane = threadIdx.x & 31;
  int tile = blockIdx.x * 8 + (threadIdx.x >> 5);   // 4096 tiles over 4096x4096
  int tm = (tile >> 6) << 6;
  int tn = (tile & 63) << 6;
  v8f acc[4][4];
  gemm64x64(X + (size_t)tm * DD, DD, W + (size_t)tn * DD, DD, DD, lane, acc);
  int rbase = (lane >> 4) << 3, cl = lane & 15;
#pragma unroll
  for (int i = 0; i < 4; i++)
#pragma unroll
    for (int j = 0; j < 4; j++)
#pragma unroll
      for (int r = 0; r < 8; r++) {
        int row = tm + i * 16 + rbase + r;          // b*L + l
        int col = tn + j * 16 + cl;                 // h*D + e
        int b = row >> 10, l = row & 1023;
        int h = col >> 9,  e = col & 511;
        size_t idx = vtrans ? ((size_t)((b * HH + h) * DD + e) * LLQ + l)
                            : ((size_t)((b * HH + h) * LLQ + l) * DD + e);
        Y[idx] = f2bf(acc[i][j][r]);
      }
}

// ---------- scores[b,h,q,k] = qh . kh^T / sqrt(D), fp32 out ----------
__global__ __launch_bounds__(256) void scores_kernel(const bf16_t* __restrict__ qh,
                                                     const bf16_t* __restrict__ kh,
                                                     float* __restrict__ sc) {
  int lane = threadIdx.x & 31;
  int tile = blockIdx.x * 8 + (threadIdx.x >> 5);   // 8192 tiles: 32 bh * 16*16
  int bh = tile >> 8;
  int t  = tile & 255;
  int tm = (t >> 4) << 6, tn = (t & 15) << 6;
  const bf16_t* A  = qh + (size_t)bh * LLQ * DD + (size_t)tm * DD;
  const bf16_t* Bm = kh + (size_t)bh * LLQ * DD + (size_t)tn * DD;
  v8f acc[4][4];
  gemm64x64(A, DD, Bm, DD, DD, lane, acc);
  const float scale = 0.044194173824159216f;        // 1/sqrt(512)
  float* out = sc + (size_t)bh * LLQ * LLQ;
  int rbase = (lane >> 4) << 3, cl = lane & 15;
#pragma unroll
  for (int i = 0; i < 4; i++)
#pragma unroll
    for (int j = 0; j < 4; j++)
#pragma unroll
      for (int r = 0; r < 8; r++)
        out[(size_t)(tm + i * 16 + rbase + r) * LLQ + (tn + j * 16 + cl)] =
            acc[i][j][r] * scale;
}

// ---------- masked softmax + renorm (allennlp semantics), bf16 attn out ----------
__global__ __launch_bounds__(256) void softmax_kernel(const float* __restrict__ sc,
                                                      const int* __restrict__ mask,
                                                      bf16_t* __restrict__ attn) {
  __shared__ float red[256];
  int tid = threadIdx.x;
  int row = blockIdx.x;                 // b*H*L + h*L + q
  int b = row >> 13;
  int q = row & 1023;
  const float* s = sc + (size_t)row * LLQ;
  const int*   m = mask + ((size_t)b * LLQ + q) * LLQ;
  float v[4], mv[4];
  float lmax = -3.0e38f;
#pragma unroll
  for (int t = 0; t < 4; t++) {
    int c = tid + t * 256;
    mv[t] = (float)m[c];
    v[t]  = s[c] * mv[t];
    lmax  = fmaxf(lmax, v[t]);
  }
  red[tid] = lmax; __syncthreads();
  for (int st = 128; st > 0; st >>= 1) { if (tid < st) red[tid] = fmaxf(red[tid], red[tid + st]); __syncthreads(); }
  float gmax = red[0]; __syncthreads();
  float lsum = 0.f;
#pragma unroll
  for (int t = 0; t < 4; t++) { v[t] = __expf(v[t] - gmax); lsum += v[t]; }
  red[tid] = lsum; __syncthreads();
  for (int st = 128; st > 0; st >>= 1) { if (tid < st) red[tid] += red[tid + st]; __syncthreads(); }
  float inv = 1.f / red[0]; __syncthreads();
  float lsum2 = 0.f;
#pragma unroll
  for (int t = 0; t < 4; t++) { v[t] = v[t] * inv * mv[t]; lsum2 += v[t]; }
  red[tid] = lsum2; __syncthreads();
  for (int st = 128; st > 0; st >>= 1) { if (tid < st) red[tid] += red[tid + st]; __syncthreads(); }
  float inv2 = 1.f / (red[0] + 1e-13f);
  bf16_t* ar = attn + (size_t)row * LLQ;
#pragma unroll
  for (int t = 0; t < 4; t++) ar[tid + t * 256] = f2bf(v[t] * inv2);
}

// ---------- attn mean over heads ----------
__global__ void mean_kernel(const bf16_t* __restrict__ attn, float* __restrict__ om) {
  size_t i = (size_t)blockIdx.x * 256 + threadIdx.x;   // over B*L*L
  int b = (int)(i >> 20);
  size_t rem = i & 0xFFFFFu;
  float s = 0.f;
#pragma unroll
  for (int h = 0; h < 8; h++) s += bf2f(attn[(((size_t)(b * 8 + h)) << 20) + rem]);
  om[i] = s * 0.125f;
}

// ---------- ctx[b,l,h*D+e] = attn . v ----------
__global__ __launch_bounds__(256) void av_kernel(const bf16_t* __restrict__ attn,
                                                 const bf16_t* __restrict__ vhT,
                                                 bf16_t* __restrict__ ctx) {
  int lane = threadIdx.x & 31;
  int tile = blockIdx.x * 8 + (threadIdx.x >> 5);     // 4096 tiles: 32 bh * 16*8
  int bh = tile >> 7;
  int t  = tile & 127;
  int tm = (t >> 3) << 6, tn = (t & 7) << 6;
  const bf16_t* A  = attn + (size_t)bh * LLQ * LLQ + (size_t)tm * LLQ;
  const bf16_t* Bm = vhT  + (size_t)bh * DD  * LLQ + (size_t)tn * LLQ;
  v8f acc[4][4];
  gemm64x64(A, LLQ, Bm, LLQ, LLQ, lane, acc);
  int b = bh >> 3, h = bh & 7;
  int rbase = (lane >> 4) << 3, cl = lane & 15;
#pragma unroll
  for (int i = 0; i < 4; i++)
#pragma unroll
    for (int j = 0; j < 4; j++)
#pragma unroll
      for (int r = 0; r < 8; r++) {
        int row = tm + i * 16 + rbase + r;
        int col = tn + j * 16 + cl;
        ctx[(size_t)(b * LLQ + row) * NDH + h * DD + col] = f2bf(acc[i][j][r]);
      }
}

// ---------- FC + bias + residual ----------
__global__ __launch_bounds__(256) void fc_kernel(const bf16_t* __restrict__ ctx,
                                                 const bf16_t* __restrict__ Wfc,
                                                 const float* __restrict__ bfc,
                                                 const float* __restrict__ resid,
                                                 float* __restrict__ xbuf) {
  int lane = threadIdx.x & 31;
  int tile = blockIdx.x * 8 + (threadIdx.x >> 5);     // 512 tiles: 64 x 8
  int tm = (tile >> 3) << 6, tn = (tile & 7) << 6;
  v8f acc[4][4];
  gemm64x64(ctx + (size_t)tm * NDH, NDH, Wfc + (size_t)tn * NDH, NDH, NDH, lane, acc);
  int rbase = (lane >> 4) << 3, cl = lane & 15;
#pragma unroll
  for (int i = 0; i < 4; i++)
#pragma unroll
    for (int j = 0; j < 4; j++)
#pragma unroll
      for (int r = 0; r < 8; r++) {
        int row = tm + i * 16 + rbase + r;
        int col = tn + j * 16 + cl;
        xbuf[(size_t)row * DD + col] =
            acc[i][j][r] + bfc[col] + resid[(size_t)row * DD + col];
      }
}

// ---------- LayerNorm (D=512, 256 threads x 2) ----------
__global__ __launch_bounds__(256) void ln_kernel(const float* __restrict__ x,
                                                 const float* __restrict__ gamma,
                                                 const float* __restrict__ beta,
                                                 float* __restrict__ out) {
  __shared__ float red[256];
  int tid = threadIdx.x, row = blockIdx.x;
  const float* xr = x + (size_t)row * DD;
  float x0 = xr[tid], x1 = xr[tid + 256];
  red[tid] = x0 + x1; __syncthreads();
  for (int st = 128; st > 0; st >>= 1) { if (tid < st) red[tid] += red[tid + st]; __syncthreads(); }
  float mu = red[0] * (1.f / 512.f); __syncthreads();
  float d0 = x0 - mu, d1 = x1 - mu;
  red[tid] = d0 * d0 + d1 * d1; __syncthreads();
  for (int st = 128; st > 0; st >>= 1) { if (tid < st) red[tid] += red[tid + st]; __syncthreads(); }
  float inv = rsqrtf(red[0] * (1.f / 512.f) + 1e-5f);
  float* o = out + (size_t)row * DD;
  o[tid]       = d0 * inv * gamma[tid]       + beta[tid];
  o[tid + 256] = d1 * inv * gamma[tid + 256] + beta[tid + 256];
}

extern "C" void kernel_launch(void* const* d_in, const int* in_sizes, int n_in,
                              void* d_out, int out_size, void* d_ws, size_t ws_size,
                              hipStream_t stream) {
  (void)in_sizes; (void)n_in; (void)out_size; (void)ws_size;
  const float* q    = (const float*)d_in[0];
  const int*   mask = (const int*)  d_in[1];
  const float* k    = (const float*)d_in[2];
  const float* v    = (const float*)d_in[3];
  const float* Wq   = (const float*)d_in[4];
  const float* Wk   = (const float*)d_in[5];
  const float* Wv   = (const float*)d_in[6];
  const float* Wfc  = (const float*)d_in[7];
  const float* bfc  = (const float*)d_in[8];
  const float* gamma= (const float*)d_in[9];
  const float* beta = (const float*)d_in[10];

  float* out       = (float*)d_out;                       // [B,L,D]
  float* attn_mean = out + (size_t)BB * LLQ * DD;         // [B,L,L]

  char* ws = (char*)d_ws;
  size_t off = 0;
  auto alloc = [&](size_t bytes) -> char* {
    char* p = ws + off;
    off = (off + bytes + 255) & ~(size_t)255;
    return p;
  };
  const int    NE  = BB * LLQ * DD;                       // 2,097,152 (also H*D*D)
  const size_t NQH = (size_t)BB * HH * LLQ * DD;          // 16,777,216

  bf16_t* qb   = (bf16_t*)alloc((size_t)NE * 2);
  bf16_t* kb   = (bf16_t*)alloc((size_t)NE * 2);
  bf16_t* vb   = (bf16_t*)alloc((size_t)NE * 2);
  bf16_t* Wqb  = (bf16_t*)alloc((size_t)NE * 2);
  bf16_t* Wkb  = (bf16_t*)alloc((size_t)NE * 2);
  bf16_t* Wvb  = (bf16_t*)alloc((size_t)NE * 2);
  bf16_t* Wfcb = (bf16_t*)alloc((size_t)NE * 2);
  bf16_t* qh   = (bf16_t*)alloc(NQH * 2);
  bf16_t* kh   = (bf16_t*)alloc(NQH * 2);
  bf16_t* vhT  = (bf16_t*)alloc(NQH * 2);
  float*  sc   = (float*) alloc((size_t)BB * HH * LLQ * LLQ * 4);
  bf16_t* attn = (bf16_t*)alloc((size_t)BB * HH * LLQ * LLQ * 2);
  // Lifetime reuse: sc is dead after softmax -> reuse for ctx;
  // qh is dead after scores -> reuse for pre-LN buffer.
  bf16_t* ctx  = (bf16_t*)sc;                             // [B,L,H*D] bf16
  float*  xbuf = (float*)qh;                              // [B*L, D] f32

  // fp32 -> bf16 conversions (all arrays are NE elements)
  cvt_kernel<<<NE / 256, 256, 0, stream>>>(q,   qb,   NE);
  cvt_kernel<<<NE / 256, 256, 0, stream>>>(k,   kb,   NE);
  cvt_kernel<<<NE / 256, 256, 0, stream>>>(v,   vb,   NE);
  cvt_kernel<<<NE / 256, 256, 0, stream>>>(Wq,  Wqb,  NE);
  cvt_kernel<<<NE / 256, 256, 0, stream>>>(Wk,  Wkb,  NE);
  cvt_kernel<<<NE / 256, 256, 0, stream>>>(Wv,  Wvb,  NE);
  cvt_kernel<<<NE / 256, 256, 0, stream>>>(Wfc, Wfcb, NE);

  // QKV projections (V written K-transposed for the NT AV GEMM)
  proj_kernel<<<512, 256, 0, stream>>>(qb, Wqb, qh, 0);
  proj_kernel<<<512, 256, 0, stream>>>(kb, Wkb, kh, 0);
  proj_kernel<<<512, 256, 0, stream>>>(vb, Wvb, vhT, 1);

  // Attention scores, masked softmax, head-mean
  scores_kernel<<<1024, 256, 0, stream>>>(qh, kh, sc);
  softmax_kernel<<<BB * HH * LLQ, 256, 0, stream>>>(sc, mask, attn);
  mean_kernel<<<(BB * LLQ * LLQ) / 256, 256, 0, stream>>>(attn, attn_mean);

  // attn @ V, FC(+bias+residual), LayerNorm
  av_kernel<<<512, 256, 0, stream>>>(attn, vhT, ctx);
  fc_kernel<<<64, 256, 0, stream>>>(ctx, Wfcb, bfc, q, xbuf);
  ln_kernel<<<BB * LLQ, 256, 0, stream>>>(xbuf, gamma, beta, out);
}
